// STAR_12068858102033
// MI455X (gfx1250) — matrix operand order repaired and checked by reading, not compiled, not run on hardware
//
#include <hip/hip_runtime.h>

// ---------------------------------------------------------------------------
// STAR fused 3-layer MLP for gfx1250 (MI455X), bf16 WMMA, f32 accumulate.
// B=65536, D=512, J=8, HIDDEN=[256,128,64].
//
// Workspace layout (d_ws):
//   [0        .. 2097152)   packed bf16 w0: J=8, NT=16, KT=16 tiles of 1KB
//   [2097152  .. 2621440)   packed bf16 w1: J=8, NT=8,  KT=8
//   [2621440  .. 2752512)   packed bf16 w2: J=8, NT=4,  KT=4
//   [2752512  .. 2760704)   fused bias0 f32 [8][256]
//   [2760704  .. 2764800)   fused bias1 f32 [8][128]
//   [2764800  .. 2766848)   fused bias2 f32 [8][64]
//   [4194304  .. 71303168)  OPTIONAL bf16 x [65536][512]  (fast path only)
// ---------------------------------------------------------------------------

typedef __attribute__((ext_vector_type(16))) __bf16 v16bf;
typedef __attribute__((ext_vector_type(8)))  float  v8f;

#define WS_W0   0
#define WS_W1   2097152
#define WS_W2   2621440
#define WS_B0   2752512
#define WS_B1   2760704
#define WS_B2   2764800
#define WS_XBF  4194304
#define WS_FAST_BYTES (4194304ull + 67108864ull)

__device__ __forceinline__ unsigned int f2bf(float f) {
  // round-to-nearest-even f32 -> bf16 bits
  unsigned int u = __float_as_uint(f);
  return (u + 0x7FFFu + ((u >> 16) & 1u)) >> 16;
}

union Frag16 { uint4 q[2]; v16bf v; };

// A operand (16x32 bf16) from LDS, row-major bf16, ldDw dwords per row.
__device__ __forceinline__ v16bf load_a(const unsigned int* base_dw, int row,
                                        int ldDw, int kt, int half) {
  const uint4* p = (const uint4*)(base_dw + row * ldDw);
  Frag16 f;
  f.q[0] = p[kt * 4 + half * 2];       // K = kt*32 + half*16 + [0..7]
  f.q[1] = p[kt * 4 + half * 2 + 1];   // K = kt*32 + half*16 + 8 + [0..7]
  return f.v;
}

// B operand (32x16 bf16) from pre-packed global tile (256 dwords, 8/lane).
__device__ __forceinline__ v16bf load_b(const unsigned int* g, int lane) {
  const uint4* p = (const uint4*)(g + lane * 8);
  Frag16 f;
  f.q[0] = p[0];
  f.q[1] = p[1];
  return f.v;
}

// ---------------------------------------------------------------------------
// Prologue 1: fuse sk*dk[j] -> bf16, swizzled into WMMA-B tile layout.
// B layout per 32x16 tile (contiguous k order, same as load_a): lane = half*16 + col holds column col;
// dword d packs K = kt*32 + half*16 + 2d (lo) and +1 (hi).
// ---------------------------------------------------------------------------
__global__ void pack_w(const float* __restrict__ sk, const float* __restrict__ dk,
                       unsigned int* __restrict__ wp, int K, int N, int KT, int NT,
                       int total) {
  int t = blockIdx.x * blockDim.x + threadIdx.x;
  if (t >= total) return;
  const int d    = t & 7;
  const int lane = (t >> 3) & 31;
  int rest = t >> 8;
  const int kt = rest % KT; rest /= KT;
  const int nt = rest % NT;
  const int j  = rest / NT;
  const int coln = lane & 15, half = lane >> 4;
  const int k = kt * 32 + half * 16 + d * 2;
  const int n = nt * 16 + coln;
  const float w0 = sk[k * N + n]       * dk[((long)j * K + k) * N + n];
  const float w1 = sk[(k + 1) * N + n] * dk[((long)j * K + k + 1) * N + n];
  wp[t] = f2bf(w0) | (f2bf(w1) << 16);
}

__global__ void pack_bias(const float* __restrict__ sb, const float* __restrict__ db,
                          float* __restrict__ bp, int N, int total) {
  int t = blockIdx.x * blockDim.x + threadIdx.x;
  if (t >= total) return;
  bp[t] = sb[t % N] + db[t];
}

// Prologue 2 (fast path): x f32 -> packed bf16, 8 elements per thread.
__global__ void cvt_x(const float* __restrict__ x, unsigned int* __restrict__ xbf) {
  const long t = (long)blockIdx.x * blockDim.x + threadIdx.x;   // 4,194,304 threads
  const float4* src = (const float4*)x;
  float4 a = src[2 * t], b = src[2 * t + 1];
  uint4 o;
  o.x = f2bf(a.x) | (f2bf(a.y) << 16);
  o.y = f2bf(a.z) | (f2bf(a.w) << 16);
  o.z = f2bf(b.x) | (f2bf(b.y) << 16);
  o.w = f2bf(b.z) | (f2bf(b.w) << 16);
  ((uint4*)xbf)[t] = o;
}

// ---------------------------------------------------------------------------
// Main fused kernel: one WG = 64 batch rows, 8 waves (wave32).
// LDS arena (dynamic, 116736 B):
//   [0      .. 65536)  x tile  bf16 [64][512]
//   [65536  .. 98304)  h0      bf16 [64][256]
//   [98304  ..114688)  h1      bf16 [64][128]
//   [114688 ..116736)  ind     f32  [64][8]
// ---------------------------------------------------------------------------
#define STAR_LDS_BYTES 116736

template <bool ASYNCX>
__global__ __launch_bounds__(256, 1) void star_fused(
    const float* __restrict__ x, const unsigned int* __restrict__ xbf,
    const float* __restrict__ ind,
    const unsigned int* __restrict__ w0, const unsigned int* __restrict__ w1,
    const unsigned int* __restrict__ w2,
    const float* __restrict__ b0, const float* __restrict__ b1,
    const float* __restrict__ b2, float* __restrict__ out) {
  extern __shared__ __align__(16) char smem[];
  unsigned int* xb = (unsigned int*)smem;                // dword view of bf16 x tile
  unsigned int* h0 = (unsigned int*)(smem + 65536);
  unsigned int* h1 = (unsigned int*)(smem + 98304);
  float*      indb = (float*)(smem + 114688);

  const int tid  = threadIdx.x;
  const int wave = tid >> 5, lane = tid & 31;
  const int half = lane >> 4, col = lane & 15;
  const long row0 = (long)blockIdx.x * 64;

  // 32-bit LDS byte offset of the arena base (for async-to-LDS destinations)
  const unsigned lds_base =
      (unsigned)(unsigned long long)(__attribute__((address_space(3))) char*)smem;

  // ---- stage indicator tile [64][8] f32 via async memory->LDS copy ----
  {
    unsigned long long ga = (unsigned long long)(const char*)(ind + row0 * 8) +
                            (unsigned)(tid * 8);
    unsigned la = lds_base + 114688u + (unsigned)(tid * 8);
    asm volatile("global_load_async_to_lds_b64 %0, %1, off"
                 :: "v"(la), "v"(ga) : "memory");
  }

  // ---- stage x tile (bf16 [64][512], 64KB) ----
  if constexpr (ASYNCX) {
    // fast path: x already bf16 in workspace -> pure async DMA, no VALU
    const char* gs = (const char*)xbf + row0 * 1024;   // 1024 B per row
#pragma unroll
    for (int i = 0; i < 16; ++i) {
      unsigned off = (unsigned)(i * 4096 + tid * 16);
      unsigned la = lds_base + off;
      unsigned long long ga = (unsigned long long)(gs + off);
      asm volatile("global_load_async_to_lds_b128 %0, %1, off"
                   :: "v"(la), "v"(ga) : "memory");
    }
  } else {
    // fallback: load f32 and convert in-kernel
    const float4* xg = (const float4*)(x + row0 * 512);
    for (int i = tid; i < 64 * 512 / 4; i += 256) {
      float4 v = xg[i];
      xb[2 * i]     = f2bf(v.x) | (f2bf(v.y) << 16);
      xb[2 * i + 1] = f2bf(v.z) | (f2bf(v.w) << 16);
    }
  }
  asm volatile("s_wait_asynccnt 0x0" ::: "memory");
  __syncthreads();

  // layer-2 / output tile ownership: wave w -> (mtA, ntO) and (mtB, ntO)
  const int mtA = wave >> 2, mtB = mtA + 2, ntO = wave & 3;
  v8f oa0 = {}, oa1 = {};

  for (int j = 0; j < 8; ++j) {
    // ================= layer 0: [64x512] x [512x256] =================
    v8f acc[4][2];
#pragma unroll
    for (int mt = 0; mt < 4; ++mt) { acc[mt][0] = {}; acc[mt][1] = {}; }
#pragma unroll 2
    for (int kt = 0; kt < 16; ++kt) {
      v16bf a[4];
#pragma unroll
      for (int mt = 0; mt < 4; ++mt) a[mt] = load_a(xb, mt * 16 + col, 256, kt, half);
      v16bf bfr[2];
#pragma unroll
      for (int u = 0; u < 2; ++u)
        bfr[u] = load_b(w0 + ((j * 16 + wave * 2 + u) * 16 + kt) * 256, lane);
#pragma unroll
      for (int mt = 0; mt < 4; ++mt)
#pragma unroll
        for (int u = 0; u < 2; ++u)
          acc[mt][u] = __builtin_amdgcn_wmma_f32_16x16x32_bf16(
              false, a[mt], false, bfr[u], (short)0, acc[mt][u], false, false);
    }
    // bias + relu -> h0 (bf16 in LDS)
#pragma unroll
    for (int u = 0; u < 2; ++u) {
      const int n = wave * 32 + u * 16 + col;
      const float bias = b0[j * 256 + n];
#pragma unroll
      for (int mt = 0; mt < 4; ++mt)
#pragma unroll
        for (int r = 0; r < 8; ++r) {
          float v = fmaxf(acc[mt][u][r] + bias, 0.0f);
          const int m = mt * 16 + r + 8 * half;
          ((unsigned short*)h0)[m * 256 + n] = (unsigned short)f2bf(v);
        }
    }
    __syncthreads();

    // ================= layer 1: [64x256] x [256x128] =================
    v8f a1[4];
#pragma unroll
    for (int mt = 0; mt < 4; ++mt) a1[mt] = {};
#pragma unroll 2
    for (int kt = 0; kt < 8; ++kt) {
      v16bf bfr = load_b(w1 + ((j * 8 + wave) * 8 + kt) * 256, lane);
#pragma unroll
      for (int mt = 0; mt < 4; ++mt) {
        v16bf a = load_a(h0, mt * 16 + col, 128, kt, half);
        a1[mt] = __builtin_amdgcn_wmma_f32_16x16x32_bf16(
            false, a, false, bfr, (short)0, a1[mt], false, false);
      }
    }
    {
      const int n = wave * 16 + col;
      const float bias = b1[j * 128 + n];
#pragma unroll
      for (int mt = 0; mt < 4; ++mt)
#pragma unroll
        for (int r = 0; r < 8; ++r) {
          float v = fmaxf(a1[mt][r] + bias, 0.0f);
          const int m = mt * 16 + r + 8 * half;
          ((unsigned short*)h1)[m * 128 + n] = (unsigned short)f2bf(v);
        }
    }
    __syncthreads();

    // ================= layer 2: [64x128] x [128x64] =================
    v8f a2a = {}, a2b = {};
#pragma unroll
    for (int kt = 0; kt < 4; ++kt) {
      v16bf bfr = load_b(w2 + ((j * 4 + ntO) * 4 + kt) * 256, lane);
      v16bf aA = load_a(h1, mtA * 16 + col, 64, kt, half);
      v16bf aB = load_a(h1, mtB * 16 + col, 64, kt, half);
      a2a = __builtin_amdgcn_wmma_f32_16x16x32_bf16(false, aA, false, bfr, (short)0, a2a, false, false);
      a2b = __builtin_amdgcn_wmma_f32_16x16x32_bf16(false, aB, false, bfr, (short)0, a2b, false, false);
    }
    const int n2 = ntO * 16 + col;
    const float bias2 = b2[j * 64 + n2];
#pragma unroll
    for (int r = 0; r < 8; ++r) {
      float va = fmaxf(a2a[r] + bias2, 0.0f);
      float vb = fmaxf(a2b[r] + bias2, 0.0f);
      oa0[r] += va * indb[(mtA * 16 + r + 8 * half) * 8 + j];
      oa1[r] += vb * indb[(mtB * 16 + r + 8 * half) * 8 + j];
    }
    // Cross-iteration LDS hazards are covered by the two barriers above.
  }

  // ---- epilogue: write out [64 x 64] tile, each element exactly once ----
  float* op = out + row0 * 64;
#pragma unroll
  for (int r = 0; r < 8; ++r) {
    op[(mtA * 16 + r + 8 * half) * 64 + ntO * 16 + col] = oa0[r];
    op[(mtB * 16 + r + 8 * half) * 64 + ntO * 16 + col] = oa1[r];
  }
}

// ---------------------------------------------------------------------------
extern "C" void kernel_launch(void* const* d_in, const int* in_sizes, int n_in,
                              void* d_out, int out_size, void* d_ws, size_t ws_size,
                              hipStream_t stream) {
  const float* x   = (const float*)d_in[0];
  const float* ind = (const float*)d_in[1];
  const float* sk0 = (const float*)d_in[2];
  const float* sb0 = (const float*)d_in[3];
  const float* dk0 = (const float*)d_in[4];
  const float* db0 = (const float*)d_in[5];
  const float* sk1 = (const float*)d_in[6];
  const float* sb1 = (const float*)d_in[7];
  const float* dk1 = (const float*)d_in[8];
  const float* db1 = (const float*)d_in[9];
  const float* sk2 = (const float*)d_in[10];
  const float* sb2 = (const float*)d_in[11];
  const float* dk2 = (const float*)d_in[12];
  const float* db2 = (const float*)d_in[13];

  unsigned char* ws = (unsigned char*)d_ws;
  unsigned int* w0p = (unsigned int*)(ws + WS_W0);
  unsigned int* w1p = (unsigned int*)(ws + WS_W1);
  unsigned int* w2p = (unsigned int*)(ws + WS_W2);
  float* b0p = (float*)(ws + WS_B0);
  float* b1p = (float*)(ws + WS_B1);
  float* b2p = (float*)(ws + WS_B2);
  unsigned int* xbf = (unsigned int*)(ws + WS_XBF);

  // fuse + pack weights (layout index == thread index)
  pack_w<<<2048, 256, 0, stream>>>(sk0, dk0, w0p, 512, 256, 16, 16, 524288);
  pack_w<<<512,  256, 0, stream>>>(sk1, dk1, w1p, 256, 128, 8,  8,  131072);
  pack_w<<<128,  256, 0, stream>>>(sk2, dk2, w2p, 128, 64,  4,  4,  32768);
  pack_bias<<<8, 256, 0, stream>>>(sb0, db0, b0p, 256, 2048);
  pack_bias<<<4, 256, 0, stream>>>(sb1, db1, b1p, 128, 1024);
  pack_bias<<<2, 256, 0, stream>>>(sb2, db2, b2p, 64, 512);

  const bool fast = (ws_size >= WS_FAST_BYTES);   // deterministic per-process
  if (fast) cvt_x<<<16384, 256, 0, stream>>>(x, xbf);

  (void)hipFuncSetAttribute((const void*)star_fused<true>,
                            hipFuncAttributeMaxDynamicSharedMemorySize,
                            STAR_LDS_BYTES);
  (void)hipFuncSetAttribute((const void*)star_fused<false>,
                            hipFuncAttributeMaxDynamicSharedMemorySize,
                            STAR_LDS_BYTES);

  if (fast)
    star_fused<true><<<1024, 256, STAR_LDS_BYTES, stream>>>(
        x, xbf, ind, w0p, w1p, w2p, b0p, b1p, b2p, (float*)d_out);
  else
    star_fused<false><<<1024, 256, STAR_LDS_BYTES, stream>>>(
        x, xbf, ind, w0p, w1p, w2p, b0p, b1p, b2p, (float*)d_out);
}
